// CausalSelfAttention_69243462746246
// MI455X (gfx1250) — matrix operand order repaired and checked by reading, not compiled, not run on hardware
//
#include <hip/hip_runtime.h>
#include <hip/hip_bf16.h>

// Problem constants (match reference)
#define DM      1024
#define TS      2048
#define NBATCH  4
#define NHEAD   16
#define HDIM    64
#define MROWS   (NBATCH * TS)   // 8192
#define N_QKV   (3 * DM)        // 3072

typedef __attribute__((ext_vector_type(8)))  float  v8f;
typedef __attribute__((ext_vector_type(16))) __bf16 v16bf;
typedef __attribute__((ext_vector_type(8)))  __bf16 v8bf;

union BF16x16 { v16bf v; v8bf h[2]; };

__device__ __forceinline__ v8f vzero8() {
    v8f z;
#pragma unroll
    for (int i = 0; i < 8; ++i) z[i] = 0.0f;
    return z;
}

__device__ __forceinline__ v8f wmma_bf16(v16bf a, v16bf b, v8f c) {
    // D = A(16x32 bf16) x B(32x16 bf16) + C(16x16 f32)
    return __builtin_amdgcn_wmma_f32_16x16x32_bf16(
        /*neg_a=*/false, a, /*neg_b=*/false, b,
        /*c_mod=*/(short)0, c, /*reuse_a=*/false, /*reuse_b=*/false);
}

// ---------------------------------------------------------------------------
// fp32 -> bf16 conversion
// ---------------------------------------------------------------------------
__global__ void f32_to_bf16_kernel(const float* __restrict__ src,
                                   unsigned short* __restrict__ dst, int n) {
    int i = blockIdx.x * blockDim.x + threadIdx.x;
    if (i < n) {
        ((__bf16*)dst)[i] = (__bf16)src[i];
    }
}

// ---------------------------------------------------------------------------
// QKV projection GEMM: qkv = x @ w_qkv^T + b_qkv, scattered into
//   Q[b,h,t,d] (pre-scaled by 1/sqrt(hd)=0.125), K[b,h,t,d], Vt[b,h,d,t]  (bf16)
// Block = 128 threads (4 waves). Block tile 128Mx64N; wave tile 32Mx64N.
// All 4 B fragments are loaded up-front each k-step (independent VGPR ranges
// -> clustered loads + partial s_wait_loadcnt), then 8 WMMAs consume them.
// ---------------------------------------------------------------------------
__global__ __launch_bounds__(128)
void qkv_gemm_kernel(const unsigned short* __restrict__ Xus,   // [8192,1024] bf16
                     const unsigned short* __restrict__ Wus,   // [3072,1024] bf16
                     const float* __restrict__ bias,           // [3072]
                     unsigned short* __restrict__ Qus,         // [B,H,T,hd]
                     unsigned short* __restrict__ Kus,         // [B,H,T,hd]
                     unsigned short* __restrict__ Vtus)        // [B,H,hd,T]
{
    const __bf16* X = (const __bf16*)Xus;
    const __bf16* W = (const __bf16*)Wus;
    __bf16* Qo = (__bf16*)Qus;
    __bf16* Ko = (__bf16*)Kus;
    __bf16* Vt = (__bf16*)Vtus;

    const int lane  = threadIdx.x & 31;
    const int wid   = threadIdx.x >> 5;
    const int khalf = lane >> 4;     // 0 or 1
    const int l16   = lane & 15;

    const int Mbase = blockIdx.y * 128 + wid * 32;   // wave owns 32 M rows
    const int n0    = blockIdx.x * 64;

    const __bf16* aptr0 = X + (size_t)(Mbase + l16) * DM;
    const __bf16* aptr1 = X + (size_t)(Mbase + 16 + l16) * DM;
    const __bf16* bbase = W + (size_t)(n0 + l16) * DM + khalf * 16;

    v8f acc[2][4];
#pragma unroll
    for (int u = 0; u < 2; ++u)
#pragma unroll
        for (int s = 0; s < 4; ++s) acc[u][s] = vzero8();

#pragma unroll 2
    for (int k0 = 0; k0 < DM; k0 += 32) {
        BF16x16 a0, a1;
        v16bf b0, b1, b2, b3;
        a0.h[0] = *(const v8bf*)(aptr0 + k0 + khalf * 16);
        a0.h[1] = *(const v8bf*)(aptr0 + k0 + khalf * 16 + 8);
        a1.h[0] = *(const v8bf*)(aptr1 + k0 + khalf * 16);
        a1.h[1] = *(const v8bf*)(aptr1 + k0 + khalf * 16 + 8);
        b0 = *(const v16bf*)(bbase + k0);
        b1 = *(const v16bf*)(bbase + (size_t)16 * DM + k0);
        b2 = *(const v16bf*)(bbase + (size_t)32 * DM + k0);
        b3 = *(const v16bf*)(bbase + (size_t)48 * DM + k0);
        if (k0 + 32 < DM) {
            __builtin_prefetch(aptr0 + k0 + 32, 0, 1);
            __builtin_prefetch(aptr1 + k0 + 32, 0, 1);
        }
        acc[0][0] = wmma_bf16(a0.v, b0, acc[0][0]);
        acc[1][0] = wmma_bf16(a1.v, b0, acc[1][0]);
        acc[0][1] = wmma_bf16(a0.v, b1, acc[0][1]);
        acc[1][1] = wmma_bf16(a1.v, b1, acc[1][1]);
        acc[0][2] = wmma_bf16(a0.v, b2, acc[0][2]);
        acc[1][2] = wmma_bf16(a1.v, b2, acc[1][2]);
        acc[0][3] = wmma_bf16(a0.v, b3, acc[0][3]);
        acc[1][3] = wmma_bf16(a1.v, b3, acc[1][3]);
    }

#pragma unroll
    for (int s = 0; s < 4; ++s) {
        const int n_sub  = n0 + s * 16;          // multiple of 16
        const float bv   = bias[n_sub + l16];
        const int region = n_sub >> 10;          // 0=Q,1=K,2=V
        const int within = n_sub & (DM - 1);
        const int h      = within >> 6;
        const int dcol   = (within & 63) + l16;  // < 64
#pragma unroll
        for (int u = 0; u < 2; ++u) {
#pragma unroll
            for (int r = 0; r < 8; ++r) {
                const int M   = Mbase + u * 16 + r + 8 * khalf;
                const int bix = M >> 11;             // / 2048
                const int t   = M & (TS - 1);
                const float v = acc[u][s][r] + bv;
                if (region == 0) {
                    Qo[(((size_t)(bix * NHEAD + h) * TS) + t) * HDIM + dcol] = (__bf16)(v * 0.125f);
                } else if (region == 1) {
                    Ko[(((size_t)(bix * NHEAD + h) * TS) + t) * HDIM + dcol] = (__bf16)v;
                } else {
                    Vt[(((size_t)(bix * NHEAD + h) * HDIM) + dcol) * TS + t] = (__bf16)v;
                }
            }
        }
    }
}

// ---------------------------------------------------------------------------
// Causal flash attention. Grid: (B*H, T/64). Block = 128 threads (4 waves).
// Wave w owns query rows [blockIdx.y*64 + 16w, +16). Keys streamed 32 at a time.
// ---------------------------------------------------------------------------
__global__ __launch_bounds__(128)
void attn_kernel(const unsigned short* __restrict__ Qus,
                 const unsigned short* __restrict__ Kus,
                 const unsigned short* __restrict__ Vtus,
                 unsigned short* __restrict__ AOus)   // [B*T, 1024] bf16
{
    const __bf16* Qg  = (const __bf16*)Qus;
    const __bf16* Kg  = (const __bf16*)Kus;
    const __bf16* Vtg = (const __bf16*)Vtus;
    __bf16* AO = (__bf16*)AOus;

    __shared__ __align__(32) __bf16 lds[4][16][32];   // P staging, per wave

    const int lane  = threadIdx.x & 31;
    const int wid   = threadIdx.x >> 5;
    const int khalf = lane >> 4;
    const int l16   = lane & 15;

    const int bh    = blockIdx.x;       // b*16 + h
    const int b     = bh >> 4;
    const int h     = bh & 15;
    const int qbase = blockIdx.y * 64 + wid * 16;

    const __bf16* Qp = Qg  + (size_t)bh * TS * HDIM;
    const __bf16* Kp = Kg  + (size_t)bh * TS * HDIM;
    const __bf16* Vp = Vtg + (size_t)bh * HDIM * TS;

    // Q A-fragments (two K=32 slabs covering hd=64), rows = qbase + (lane&15)
    BF16x16 qa0, qa1;
    {
        const __bf16* qrow = Qp + (size_t)(qbase + l16) * HDIM;
        qa0.h[0] = *(const v8bf*)(qrow + khalf * 16);
        qa0.h[1] = *(const v8bf*)(qrow + khalf * 16 + 8);
        qa1.h[0] = *(const v8bf*)(qrow + 32 + khalf * 16);
        qa1.h[1] = *(const v8bf*)(qrow + 32 + khalf * 16 + 8);
    }

    v8f o0 = vzero8(), o1 = vzero8(), o2 = vzero8(), o3 = vzero8();
    float m[8], l[8];
#pragma unroll
    for (int r = 0; r < 8; ++r) { m[r] = -1.0e30f; l[r] = 0.0f; }

    for (int kb = 0; kb < qbase + 16; kb += 32) {
        // --- S = Q @ K^T for 32 keys (two 16-key subtiles) ---
        v16bf k0lo = *(const v16bf*)(Kp + (size_t)(kb + l16) * HDIM + khalf * 16);
        v16bf k0hi = *(const v16bf*)(Kp + (size_t)(kb + l16) * HDIM + 32 + khalf * 16);
        v16bf k1lo = *(const v16bf*)(Kp + (size_t)(kb + 16 + l16) * HDIM + khalf * 16);
        v16bf k1hi = *(const v16bf*)(Kp + (size_t)(kb + 16 + l16) * HDIM + 32 + khalf * 16);

        v8f s0 = wmma_bf16(qa0.v, k0lo, vzero8());
        s0     = wmma_bf16(qa1.v, k0hi, s0);
        v8f s1 = wmma_bf16(qa0.v, k1lo, vzero8());
        s1     = wmma_bf16(qa1.v, k1hi, s1);

        // --- online softmax update + write P (bf16) to per-wave LDS tile ---
#pragma unroll
        for (int r = 0; r < 8; ++r) {
            const int row = qbase + r + 8 * khalf;
            const int c0  = kb + l16;
            const int c1  = kb + 16 + l16;
            float x0 = (c0 <= row) ? s0[r] : -1.0e30f;
            float x1 = (c1 <= row) ? s1[r] : -1.0e30f;

            float mx = fmaxf(x0, x1);
            mx = fmaxf(mx, __shfl_xor(mx, 1, 16));
            mx = fmaxf(mx, __shfl_xor(mx, 2, 16));
            mx = fmaxf(mx, __shfl_xor(mx, 4, 16));
            mx = fmaxf(mx, __shfl_xor(mx, 8, 16));
            const float newm = fmaxf(m[r], mx);
            const float sc   = __expf(m[r] - newm);
            const float p0   = __expf(x0 - newm);
            const float p1   = __expf(x1 - newm);
            float rs = p0 + p1;
            rs += __shfl_xor(rs, 1, 16);
            rs += __shfl_xor(rs, 2, 16);
            rs += __shfl_xor(rs, 4, 16);
            rs += __shfl_xor(rs, 8, 16);
            l[r] = l[r] * sc + rs;
            m[r] = newm;
            o0[r] *= sc; o1[r] *= sc; o2[r] *= sc; o3[r] *= sc;
            lds[wid][r + 8 * khalf][l16]      = (__bf16)p0;
            lds[wid][r + 8 * khalf][16 + l16] = (__bf16)p1;
        }

        // LDS stores -> transposed re-read as A-fragment (same-wave cross-lane)
        asm volatile("s_wait_dscnt 0" ::: "memory");

        BF16x16 pa;
        pa.h[0] = *(const v8bf*)(&lds[wid][l16][khalf * 16]);
        pa.h[1] = *(const v8bf*)(&lds[wid][l16][khalf * 16 + 8]);

        // --- O += P @ V  (V stored transposed: contiguous over keys) ---
        const __bf16* vb = Vp + kb + khalf * 16;
        v16bf vb0 = *(const v16bf*)(vb + (size_t)(l16) * TS);
        v16bf vb1 = *(const v16bf*)(vb + (size_t)(16 + l16) * TS);
        v16bf vb2 = *(const v16bf*)(vb + (size_t)(32 + l16) * TS);
        v16bf vb3 = *(const v16bf*)(vb + (size_t)(48 + l16) * TS);
        o0 = wmma_bf16(pa.v, vb0, o0);
        o1 = wmma_bf16(pa.v, vb1, o1);
        o2 = wmma_bf16(pa.v, vb2, o2);
        o3 = wmma_bf16(pa.v, vb3, o3);
    }

    // --- epilogue: normalize and store to [B*T, D] bf16 for out-proj GEMM ---
#pragma unroll
    for (int r = 0; r < 8; ++r) {
        const int t = qbase + r + 8 * khalf;
        const float inv = 1.0f / l[r];
        const size_t base = ((size_t)(b * TS + t)) * DM + h * HDIM;
        AO[base + l16]      = (__bf16)(o0[r] * inv);
        AO[base + 16 + l16] = (__bf16)(o1[r] * inv);
        AO[base + 32 + l16] = (__bf16)(o2[r] * inv);
        AO[base + 48 + l16] = (__bf16)(o3[r] * inv);
    }
}

// ---------------------------------------------------------------------------
// Output projection: out = AO(bf16) @ w_out^T(bf16) + b_out  -> fp32
// Wave tile 32Mx64N, block tile 128Mx64N. B fragments pre-loaded per k-step.
// ---------------------------------------------------------------------------
__global__ __launch_bounds__(128)
void out_gemm_kernel(const unsigned short* __restrict__ Aus,  // [8192,1024] bf16
                     const unsigned short* __restrict__ Wus,  // [1024,1024] bf16
                     const float* __restrict__ bias,          // [1024]
                     float* __restrict__ C)                   // [8192,1024] f32
{
    const __bf16* A = (const __bf16*)Aus;
    const __bf16* W = (const __bf16*)Wus;

    const int lane  = threadIdx.x & 31;
    const int wid   = threadIdx.x >> 5;
    const int khalf = lane >> 4;
    const int l16   = lane & 15;

    const int Mbase = blockIdx.y * 128 + wid * 32;
    const int n0    = blockIdx.x * 64;

    const __bf16* aptr0 = A + (size_t)(Mbase + l16) * DM;
    const __bf16* aptr1 = A + (size_t)(Mbase + 16 + l16) * DM;
    const __bf16* bbase = W + (size_t)(n0 + l16) * DM + khalf * 16;

    v8f acc[2][4];
#pragma unroll
    for (int u = 0; u < 2; ++u)
#pragma unroll
        for (int s = 0; s < 4; ++s) acc[u][s] = vzero8();

#pragma unroll 2
    for (int k0 = 0; k0 < DM; k0 += 32) {
        BF16x16 a0, a1;
        v16bf b0, b1, b2, b3;
        a0.h[0] = *(const v8bf*)(aptr0 + k0 + khalf * 16);
        a0.h[1] = *(const v8bf*)(aptr0 + k0 + khalf * 16 + 8);
        a1.h[0] = *(const v8bf*)(aptr1 + k0 + khalf * 16);
        a1.h[1] = *(const v8bf*)(aptr1 + k0 + khalf * 16 + 8);
        b0 = *(const v16bf*)(bbase + k0);
        b1 = *(const v16bf*)(bbase + (size_t)16 * DM + k0);
        b2 = *(const v16bf*)(bbase + (size_t)32 * DM + k0);
        b3 = *(const v16bf*)(bbase + (size_t)48 * DM + k0);
        if (k0 + 32 < DM) {
            __builtin_prefetch(aptr0 + k0 + 32, 0, 1);
            __builtin_prefetch(aptr1 + k0 + 32, 0, 1);
        }
        acc[0][0] = wmma_bf16(a0.v, b0, acc[0][0]);
        acc[1][0] = wmma_bf16(a1.v, b0, acc[1][0]);
        acc[0][1] = wmma_bf16(a0.v, b1, acc[0][1]);
        acc[1][1] = wmma_bf16(a1.v, b1, acc[1][1]);
        acc[0][2] = wmma_bf16(a0.v, b2, acc[0][2]);
        acc[1][2] = wmma_bf16(a1.v, b2, acc[1][2]);
        acc[0][3] = wmma_bf16(a0.v, b3, acc[0][3]);
        acc[1][3] = wmma_bf16(a1.v, b3, acc[1][3]);
    }

#pragma unroll
    for (int s = 0; s < 4; ++s) {
        const int n    = n0 + s * 16 + l16;
        const float bv = bias[n];
#pragma unroll
        for (int u = 0; u < 2; ++u) {
#pragma unroll
            for (int r = 0; r < 8; ++r) {
                const int M = Mbase + u * 16 + r + 8 * khalf;
                C[(size_t)M * DM + n] = acc[u][s][r] + bv;
            }
        }
    }
}

// ---------------------------------------------------------------------------
extern "C" void kernel_launch(void* const* d_in, const int* in_sizes, int n_in,
                              void* d_out, int out_size, void* d_ws, size_t ws_size,
                              hipStream_t stream) {
    const float* x     = (const float*)d_in[0];   // [4,2048,1024]
    const float* w_qkv = (const float*)d_in[1];   // [3072,1024]
    const float* b_qkv = (const float*)d_in[2];   // [3072]
    const float* w_out = (const float*)d_in[3];   // [1024,1024]
    const float* b_out = (const float*)d_in[4];   // [1024]
    float* out = (float*)d_out;                   // [4,2048,1024]

    // Workspace layout (bf16 buffers), ~92 MB total
    size_t off = 0;
    char* ws = (char*)d_ws;
    auto take = [&](size_t elems) {
        unsigned short* p = (unsigned short*)(ws + off);
        off += elems * sizeof(unsigned short);
        return p;
    };
    unsigned short* xb    = take((size_t)MROWS * DM);   // x in bf16
    unsigned short* wqkvb = take((size_t)N_QKV * DM);
    unsigned short* woutb = take((size_t)DM * DM);
    unsigned short* Qb    = take((size_t)MROWS * DM);   // [B,H,T,hd]
    unsigned short* Kb    = take((size_t)MROWS * DM);   // [B,H,T,hd]
    unsigned short* Vtb   = take((size_t)MROWS * DM);   // [B,H,hd,T]
    unsigned short* aob   = take((size_t)MROWS * DM);   // attn out bf16
    (void)ws_size; (void)in_sizes; (void)n_in; (void)out_size;

    // 1) fp32 -> bf16 conversions
    {
        int n1 = MROWS * DM;
        f32_to_bf16_kernel<<<(n1 + 255) / 256, 256, 0, stream>>>(x, xb, n1);
        int n2 = N_QKV * DM;
        f32_to_bf16_kernel<<<(n2 + 255) / 256, 256, 0, stream>>>(w_qkv, wqkvb, n2);
        int n3 = DM * DM;
        f32_to_bf16_kernel<<<(n3 + 255) / 256, 256, 0, stream>>>(w_out, woutb, n3);
    }

    // 2) QKV projection (M=8192, N=3072, K=1024), scatter to Q/K/Vt
    {
        dim3 grid(N_QKV / 64, MROWS / 128);  // (48, 64)
        qkv_gemm_kernel<<<grid, 128, 0, stream>>>(xb, wqkvb, b_qkv, Qb, Kb, Vtb);
    }

    // 3) causal flash attention
    {
        dim3 grid(NBATCH * NHEAD, TS / 64);  // (64, 32)
        attn_kernel<<<grid, 128, 0, stream>>>(Qb, Kb, Vtb, aob);
    }

    // 4) output projection (M=8192, N=1024, K=1024) -> fp32 d_out
    {
        dim3 grid(DM / 64, MROWS / 128);     // (16, 64)
        out_gemm_kernel<<<grid, 128, 0, stream>>>(aob, woutb, b_out, out);
    }
}
